// RelationalTransformer_55800215109740
// MI455X (gfx1250) — matrix operand order repaired and checked by reading, not compiled, not run on hardware
//
#include <hip/hip_runtime.h>

// ---------------------------------------------------------------------------
// RelationalTransformer forward for MI455X (gfx1250), bf16 WMMA everywhere.
//
// Input index map (assumes jax pytree flatten: params dict sorted by key,
// layers list in order, then remaining setup_inputs() entries in order):
//   0 bool_emb(2,512) 1 cat_b 2 cat_w(384,512) 3 col_b 4 col_w(384,512)
//   5 dec_bool_b 6 dec_bool_w(512,1) 7 dec_cat_b 8 dec_cat_w(512,512)
//   9 dec_null_b 10 dec_null_w 11 dec_num_b 12 dec_num_w
//   13 dec_ts_b(8) 14 dec_ts_w(512,8) 15 final_norm 16 h0_norm 17 id_emb
//   18 + 20*l + {0:b1 1:b2 2:col_norm 3:col_wk 4:col_wo 5:col_wq 6:col_wv
//                7:ffn_norm 8:in_norm 9:in_wk 10:in_wo 11:in_wq 12:in_wv
//                13:out_norm 14:out_wk 15:out_wo 16:out_wq 17:out_wv
//                18:w1(512,2048) 19:w2(2048,512)}   for l in 0..3
//   98 mask_emb 99 null_emb 100 num_b 101 num_w(1,512)
//   102 text_b 103 text_w(384,512) 104 ts_b 105 ts_w(8,512)
//   106 seq_row_ids 107 fk_adj 108 is_padding 109 column_ids
//   110 semantic_types 111 is_null 112 is_target 113 numeric_values
//   114 timestamp_values 115 bool_values 116 categorical_embed_ids
//   117 text_embed_ids 118 text_batch_embeddings(256,384)
//   119 col_emb_table(64,384) 120 cat_emb_table(10000,384)
// ---------------------------------------------------------------------------

typedef __attribute__((ext_vector_type(16))) __bf16 v16bf;
typedef __attribute__((ext_vector_type(8)))  float  v8f;

union Frag {
    v16bf v;
    uint4 q[2];
    unsigned short us[16];
};

__device__ __forceinline__ unsigned short f2bf(float f) {
    unsigned int u = __float_as_uint(f);
    unsigned int r = u + 0x7fffu + ((u >> 16) & 1u);
    return (unsigned short)(r >> 16);
}

__device__ __forceinline__ v8f zero8() {
    v8f z = {0.f, 0.f, 0.f, 0.f, 0.f, 0.f, 0.f, 0.f};
    return z;
}

// CDNA5 async global->LDS path (ASYNCcnt-tracked, bypasses VGPRs).
#if defined(__has_builtin)
#if __has_builtin(__builtin_amdgcn_global_load_async_to_lds_b128) && \
    __has_builtin(__builtin_amdgcn_s_wait_asynccnt)
#define USE_ASYNC_LDS 1
#endif
#endif

#define AS_GLOBAL __attribute__((address_space(1)))
#define AS_LOCAL  __attribute__((address_space(3)))

// 16-byte int vector type matching the async-load builtin's pointee type
typedef int v4i_vs __attribute__((vector_size(16)));

// constants
#define MB (4096)      // B*S tokens
#define DD (512)
#define SS (1024)
#define BB (4)
#define HH (8)
#define DHD (64)
#define DFF (2048)

// compile-time GEMM epilogue flags
#define FLAG_GELU  1
#define FLAG_RES   2   // f32 accumulate into Cf32
#define FLAG_F32   4   // write/accumulate f32 Cf32
#define FLAG_BF16  8   // write bf16 Cbf
#define FLAG_BIAS 16

// ---------------------------------------------------------------------------
// Generic bf16 GEMM: C[M,N] = act(A[M,K] @ W[K,N] + bias).  128x128 workgroup
// tile, 8 waves, each wave = 32x64 via 2x4 v_wmma_f32_16x16x32_bf16 tiles.
// A tile is streamed into LDS with global_load_async_to_lds_b128 when the
// toolchain exposes it; W tile goes through VGPRs because it is transposed
// on the way into LDS ([col][k]) so B-fragments are contiguous b128 reads.
// ---------------------------------------------------------------------------
template <int FLAGS>
__global__ __launch_bounds__(256) void gemm_bf16_t(
    const unsigned short* __restrict__ A,   // M x K bf16
    const unsigned short* __restrict__ W,   // K x N bf16
    const float* __restrict__ bias,         // N (FLAG_BIAS)
    float* __restrict__ Cf32,               // M x N f32 (FLAG_F32)
    unsigned short* __restrict__ Cbf,       // M x N bf16 (FLAG_BF16)
    int M, int N, int K)
{
    __shared__ unsigned short lds_a[128 * 40];   // rows padded to 40 halves
    __shared__ unsigned short lds_w[128 * 40];   // stored [col][k], padded

    const int m0 = blockIdx.x * 128;
    const int n0 = blockIdx.y * 128;
    const int t  = threadIdx.x;
    const int lane = t & 31;
    const int wid  = t >> 5;
    const int wm = wid >> 1;       // 0..3 (M sub-block)
    const int wn = wid & 1;        // 0..1 (N sub-block)
    const int hl = lane >> 4;      // half-wave select
    const int ln = lane & 15;

    v8f acc[2][4];
#pragma unroll
    for (int i = 0; i < 2; i++)
#pragma unroll
        for (int j = 0; j < 4; j++) acc[i][j] = zero8();

    for (int k0 = 0; k0 < K; k0 += 32) {
        // stage A tile (128 x 32 halves): 2x b128 per thread
        {
            int row = t >> 1;
            int off = (t & 1) * 16;
            const unsigned short* src = A + (size_t)(m0 + row) * K + k0 + off;
#ifdef USE_ASYNC_LDS
            __builtin_amdgcn_global_load_async_to_lds_b128(
                (AS_GLOBAL v4i_vs*)src,
                (AS_LOCAL v4i_vs*)&lds_a[row * 40 + off], 0, 0);
            __builtin_amdgcn_global_load_async_to_lds_b128(
                (AS_GLOBAL v4i_vs*)(src + 8),
                (AS_LOCAL v4i_vs*)&lds_a[row * 40 + off + 8], 0, 0);
#else
            const uint4* s4 = reinterpret_cast<const uint4*>(src);
            uint4 d0 = s4[0];
            uint4 d1 = s4[1];
            uint4* dst = reinterpret_cast<uint4*>(&lds_a[row * 40 + off]);
            dst[0] = d0;
            dst[1] = d1;
#endif
            if (k0 + 32 < K)  // pull next K tile toward the caches
                __builtin_prefetch(src + 32, 0, 1);
        }
        // stage W tile transposed: lds_w[col][k]
        {
            int kk = t >> 3;            // 0..31
            int ng = (t & 7) * 16;      // col group base
            const unsigned short* srow = W + (size_t)(k0 + kk) * N + n0 + ng;
#pragma unroll
            for (int j = 0; j < 16; j++)
                lds_w[(ng + j) * 40 + kk] = srow[j];
        }
#ifdef USE_ASYNC_LDS
        __builtin_amdgcn_s_wait_asynccnt(0);
#endif
        __syncthreads();

        Frag af[2];
#pragma unroll
        for (int mt = 0; mt < 2; mt++) {
            int row = wm * 32 + mt * 16 + ln;
            const uint4* p0 = reinterpret_cast<const uint4*>(&lds_a[row * 40 + 16 * hl]);
            const uint4* p1 = reinterpret_cast<const uint4*>(&lds_a[row * 40 + 16 * hl + 8]);
            af[mt].q[0] = p0[0];
            af[mt].q[1] = p1[0];
        }
        Frag bfr[4];
#pragma unroll
        for (int nt = 0; nt < 4; nt++) {
            int col = wn * 64 + nt * 16 + ln;
            const uint4* p = reinterpret_cast<const uint4*>(&lds_w[col * 40 + 16 * hl]);
            bfr[nt].q[0] = p[0];
            bfr[nt].q[1] = p[1];
        }
#pragma unroll
        for (int mt = 0; mt < 2; mt++)
#pragma unroll
            for (int nt = 0; nt < 4; nt++)
                acc[mt][nt] = __builtin_amdgcn_wmma_f32_16x16x32_bf16(
                    false, af[mt].v, false, bfr[nt].v, (short)0, acc[mt][nt],
                    false, false);
        __syncthreads();
    }

    // straight-line epilogue (all decisions compile-time)
#pragma unroll
    for (int mt = 0; mt < 2; mt++) {
#pragma unroll
        for (int nt = 0; nt < 4; nt++) {
            int col = n0 + wn * 64 + nt * 16 + ln;
            float bv = (FLAGS & FLAG_BIAS) ? bias[col] : 0.0f;
#pragma unroll
            for (int r = 0; r < 8; r++) {
                int row = m0 + wm * 32 + mt * 16 + r + 8 * hl;
                float v = acc[mt][nt][r] + bv;
                if (FLAGS & FLAG_GELU) {
                    float x = v;
                    float x3 = x * x * x;
                    v = 0.5f * x * (1.0f + tanhf(0.7978845608028654f * (x + 0.044715f * x3)));
                }
                size_t idx = (size_t)row * N + col;
                if (FLAGS & FLAG_F32) {
                    if (FLAGS & FLAG_RES) Cf32[idx] += v;
                    else                  Cf32[idx]  = v;
                }
                if (FLAGS & FLAG_BF16) Cbf[idx] = f2bf(v);
            }
        }
    }
}

// ---------------------------------------------------------------------------
// Flash-style masked attention.  One wave handles (b, head, 16-row q tile);
// inner loop walks 32-key chunks: 4 WMMA for QK^T, online softmax, P through
// LDS (C-layout -> A-layout transpose), 4 WMMA for P*V against Vt[B,H,DH,S].
// ---------------------------------------------------------------------------
__global__ __launch_bounds__(32) void attention_kernel(
    const unsigned short* __restrict__ Q,    // [B,S,D] bf16
    const unsigned short* __restrict__ Km,   // [B,S,D] bf16
    const unsigned short* __restrict__ Vt,   // [B,H,DH,S] bf16
    const unsigned char* __restrict__ mask,  // [B,S,S]
    unsigned short* __restrict__ O)          // [B,S,D] bf16
{
    __shared__ unsigned short lds_p[16 * 32];

    const int bid = blockIdx.x;
    const int qt = bid & 63;
    const int hh = (bid >> 6) & 7;
    const int b  = bid >> 9;
    const int lane = threadIdx.x;
    const int hl = lane >> 4;
    const int ln = lane & 15;
    const int q0 = qt * 16;

    // Q A-fragments (16x64 split into two K=32 chunks)
    Frag qa[2];
#pragma unroll
    for (int c = 0; c < 2; c++) {
        int row = q0 + ln;
        const unsigned short* base = Q + (size_t)(b * SS + row) * DD + hh * DHD + c * 32;
        const uint4* p0 = reinterpret_cast<const uint4*>(base + 16 * hl);
        const uint4* p1 = reinterpret_cast<const uint4*>(base + 16 * hl + 8);
        qa[c].q[0] = p0[0];
        qa[c].q[1] = p1[0];
    }

    v8f oacc[4];
#pragma unroll
    for (int i = 0; i < 4; i++) oacc[i] = zero8();
    float rm[8], rs[8];
#pragma unroll
    for (int r = 0; r < 8; r++) { rm[r] = -1e30f; rs[r] = 0.f; }

    for (int kc = 0; kc < 32; kc++) {
        const int key0 = kc * 32;
        v8f sc[2];
        sc[0] = zero8();
        sc[1] = zero8();
#pragma unroll
        for (int st = 0; st < 2; st++) {
#pragma unroll
            for (int c = 0; c < 2; c++) {
                Frag kb;
                int key = key0 + st * 16 + ln;
                const unsigned short* base =
                    Km + (size_t)(b * SS + key) * DD + hh * DHD + c * 32 + 16 * hl;
                const uint4* p = reinterpret_cast<const uint4*>(base);
                kb.q[0] = p[0];
                kb.q[1] = p[1];
                sc[st] = __builtin_amdgcn_wmma_f32_16x16x32_bf16(
                    false, qa[c].v, false, kb.v, (short)0, sc[st], false, false);
            }
        }

        // masked scores + online softmax (rows r live across 16-lane halves)
        float s0[8], s1[8], p0v[8], p1v[8];
#pragma unroll
        for (int r = 0; r < 8; r++) {
            int qrow = q0 + r + 8 * hl;
            const unsigned char* mrow = mask + (size_t)(b * SS + qrow) * SS;
            bool ma = mrow[key0 + ln] != 0;
            bool mbv = mrow[key0 + 16 + ln] != 0;
            s0[r] = ma  ? sc[0][r] * 0.125f : -1e30f;
            s1[r] = mbv ? sc[1][r] * 0.125f : -1e30f;
        }
#pragma unroll
        for (int r = 0; r < 8; r++) {
            float cm = fmaxf(s0[r], s1[r]);
#pragma unroll
            for (int m = 1; m < 16; m <<= 1) cm = fmaxf(cm, __shfl_xor(cm, m, 32));
            float mnew = fmaxf(rm[r], cm);
            float scale = __expf(rm[r] - mnew);
            float p0 = (s0[r] > -1e29f) ? __expf(s0[r] - mnew) : 0.f;
            float p1 = (s1[r] > -1e29f) ? __expf(s1[r] - mnew) : 0.f;
            float ls = p0 + p1;
#pragma unroll
            for (int m = 1; m < 16; m <<= 1) ls += __shfl_xor(ls, m, 32);
            rs[r] = rs[r] * scale + ls;
            rm[r] = mnew;
#pragma unroll
            for (int nt = 0; nt < 4; nt++) oacc[nt][r] *= scale;
            p0v[r] = p0;
            p1v[r] = p1;
        }

        // P: C-layout -> A-layout via LDS
#pragma unroll
        for (int r = 0; r < 8; r++) {
            int prow = r + 8 * hl;
            lds_p[prow * 32 + ln]      = f2bf(p0v[r]);
            lds_p[prow * 32 + 16 + ln] = f2bf(p1v[r]);
        }
        __syncthreads();
        Frag pa;
        {
            const uint4* p0 = reinterpret_cast<const uint4*>(&lds_p[ln * 32 + 16 * hl]);
            const uint4* p1 = reinterpret_cast<const uint4*>(&lds_p[ln * 32 + 16 * hl + 8]);
            pa.q[0] = p0[0];
            pa.q[1] = p1[0];
        }
        __syncthreads();

        // O += P(16x32) * V(32x64)
#pragma unroll
        for (int nt = 0; nt < 4; nt++) {
            Frag vb;
            int dcol = nt * 16 + ln;
            const unsigned short* base =
                Vt + ((size_t)(b * HH + hh) * DHD + dcol) * SS + key0 + 16 * hl;
            const uint4* p = reinterpret_cast<const uint4*>(base);
            vb.q[0] = p[0];
            vb.q[1] = p[1];
            oacc[nt] = __builtin_amdgcn_wmma_f32_16x16x32_bf16(
                false, pa.v, false, vb.v, (short)0, oacc[nt], false, false);
        }
    }

#pragma unroll
    for (int nt = 0; nt < 4; nt++) {
        int col = hh * DHD + nt * 16 + ln;
#pragma unroll
        for (int r = 0; r < 8; r++) {
            int row = q0 + r + 8 * hl;
            float v = (rs[r] > 0.f) ? (oacc[nt][r] / rs[r]) : 0.f;
            O[(size_t)(b * SS + row) * DD + col] = f2bf(v);
        }
    }
}

// ---------------------------------------------------------------------------
// Value encode: embedding gathers + DT=384 projections + RMS, one token/block.
// ---------------------------------------------------------------------------
__global__ __launch_bounds__(128) void encode_kernel(
    const int* __restrict__ column_ids, const int* __restrict__ semantic_types,
    const int* __restrict__ is_null, const int* __restrict__ is_target,
    const int* __restrict__ is_padding,
    const float* __restrict__ numeric_values, const float* __restrict__ ts_values,
    const int* __restrict__ bool_values, const int* __restrict__ cat_ids,
    const int* __restrict__ text_ids,
    const float* __restrict__ text_emb, const float* __restrict__ col_table,
    const float* __restrict__ cat_table,
    const float* __restrict__ col_w, const float* __restrict__ col_b,
    const float* __restrict__ id_emb,
    const float* __restrict__ num_w, const float* __restrict__ num_b,
    const float* __restrict__ ts_w, const float* __restrict__ ts_b,
    const float* __restrict__ bool_emb,
    const float* __restrict__ cat_w, const float* __restrict__ cat_b,
    const float* __restrict__ text_w, const float* __restrict__ text_b,
    const float* __restrict__ null_emb, const float* __restrict__ mask_emb,
    const float* __restrict__ h0_norm,
    float* __restrict__ h)
{
    __shared__ float red[128];
    const int tok = blockIdx.x;
    const int t = threadIdx.x;
    const int cid = column_ids[tok];
    const int st  = semantic_types[tok];
    const int isn = is_null[tok];
    const int ist = is_target[tok];
    const int pad = is_padding[tok];
    const float numv = numeric_values[tok];
    const int bv = bool_values[tok];
    const int catid = cat_ids[tok];
    int tid = text_ids[tok];
    tid = tid < 0 ? 0 : (tid > 255 ? 255 : tid);

    const float* crow  = col_table + (size_t)cid * 384;
    const float* carow = cat_table + (size_t)catid * 384;
    const float* trow  = text_emb + (size_t)tid * 384;

    float x[4];
    float ss = 0.f;
#pragma unroll
    for (int i = 0; i < 4; i++) {
        int d = t + 128 * i;
        float ce = col_b[d];
        for (int j = 0; j < 384; j++) ce += crow[j] * col_w[j * 512 + d];
        float raw;
        if (st == 0) raw = id_emb[d];
        else if (st == 1) raw = numv * num_w[d] + num_b[d];
        else if (st == 2) {
            raw = ts_b[d];
            for (int j = 0; j < 8; j++) raw += ts_values[tok * 8 + j] * ts_w[j * 512 + d];
        } else if (st == 3) raw = bool_emb[bv * 512 + d];
        else if (st == 4) {
            raw = cat_b[d];
            for (int j = 0; j < 384; j++) raw += carow[j] * cat_w[j * 512 + d];
        } else if (st == 5) {
            raw = text_b[d];
            for (int j = 0; j < 384; j++) raw += trow[j] * text_w[j * 512 + d];
        } else raw = 0.f;
        float v  = isn ? null_emb[d] : raw;
        float vf = ist ? mask_emb[d] : v;
        x[i] = ce + vf;
        ss += x[i] * x[i];
    }
    red[t] = ss;
    __syncthreads();
    for (int o = 64; o > 0; o >>= 1) {
        if (t < o) red[t] += red[t + o];
        __syncthreads();
    }
    float inv = rsqrtf(red[0] / 512.f + 1e-6f);
#pragma unroll
    for (int i = 0; i < 4; i++) {
        int d = t + 128 * i;
        float v = x[i] * inv * (1.f + h0_norm[d]);
        h[(size_t)tok * 512 + d] = pad ? 0.f : v;
    }
}

// ---------------------------------------------------------------------------
__global__ void build_masks_kernel(
    const int* __restrict__ rowids, const int* __restrict__ fk,
    const int* __restrict__ pad, const int* __restrict__ colids,
    unsigned char* __restrict__ mo, unsigned char* __restrict__ mi,
    unsigned char* __restrict__ mc)
{
    size_t idx = (size_t)blockIdx.x * 256 + threadIdx.x;
    int j = idx & 1023;
    int i = (idx >> 10) & 1023;
    int b = (int)(idx >> 20);
    int ri = rowids[b * 1024 + i];
    int rj = rowids[b * 1024 + j];
    int fij = fk[(b * 64 + ri) * 64 + rj];
    int fji = fk[(b * 64 + rj) * 64 + ri];
    int valid = (!pad[b * 1024 + i]) && (!pad[b * 1024 + j]);
    mo[idx] = (unsigned char)((((ri == rj) || fij) && valid) ? 1 : 0);
    mi[idx] = (unsigned char)((fji && valid) ? 1 : 0);
    mc[idx] = (unsigned char)(((colids[b * 1024 + i] == colids[b * 1024 + j]) && valid) ? 1 : 0);
}

// ---------------------------------------------------------------------------
__global__ __launch_bounds__(128) void rms_cast_kernel(
    const float* __restrict__ h, const float* __restrict__ w,
    unsigned short* __restrict__ Xbf, float* __restrict__ Xf32)
{
    __shared__ float red[128];
    const int tok = blockIdx.x;
    const int t = threadIdx.x;
    const float* row = h + (size_t)tok * 512;
    float x[4];
    float ss = 0.f;
#pragma unroll
    for (int i = 0; i < 4; i++) {
        x[i] = row[t + 128 * i];
        ss += x[i] * x[i];
    }
    red[t] = ss;
    __syncthreads();
    for (int o = 64; o > 0; o >>= 1) {
        if (t < o) red[t] += red[t + o];
        __syncthreads();
    }
    float inv = rsqrtf(red[0] / 512.f + 1e-6f);
#pragma unroll
    for (int i = 0; i < 4; i++) {
        int d = t + 128 * i;
        float v = x[i] * inv * (1.f + w[d]);
        Xbf[(size_t)tok * 512 + d] = f2bf(v);
        if (Xf32) Xf32[(size_t)tok * 512 + d] = v;
    }
}

// ---------------------------------------------------------------------------
__global__ void cast_bf16_kernel(const float* __restrict__ src,
                                 unsigned short* __restrict__ dst, int n)
{
    int i = blockIdx.x * 256 + threadIdx.x;
    if (i < n) dst[i] = f2bf(src[i]);
}

// ---------------------------------------------------------------------------
__global__ void transpose_v_kernel(const unsigned short* __restrict__ V,
                                   unsigned short* __restrict__ Vt)
{
    int idx = blockIdx.x * 256 + threadIdx.x;   // B*S*D = 2097152
    int d = idx & 511;
    int s = (idx >> 9) & 1023;
    int b = idx >> 19;
    int hh = d >> 6;
    int dh = d & 63;
    Vt[((size_t)(b * HH + hh) * DHD + dh) * SS + s] = V[idx];
}

// ---------------------------------------------------------------------------
__global__ __launch_bounds__(64) void decoders_kernel(
    const float* __restrict__ hf,
    const float* __restrict__ wnull, const float* __restrict__ bnull,
    const float* __restrict__ wnum, const float* __restrict__ bnum,
    const float* __restrict__ wbool, const float* __restrict__ bbool,
    const float* __restrict__ wts, const float* __restrict__ bts,
    float* __restrict__ onull, float* __restrict__ onum,
    float* __restrict__ obool, float* __restrict__ ots)
{
    __shared__ float red[256];
    const int tok = blockIdx.x;
    const int t = threadIdx.x;
    const float* row = hf + (size_t)tok * 512;
    float a0 = 0.f, a1 = 0.f, a2 = 0.f;
    for (int d = t; d < 512; d += 64) {
        float x = row[d];
        a0 += x * wnull[d];
        a1 += x * wnum[d];
        a2 += x * wbool[d];
    }
    red[t] = a0;
    red[64 + t] = a1;
    red[128 + t] = a2;
    __syncthreads();
    for (int o = 32; o > 0; o >>= 1) {
        if (t < o) {
            red[t] += red[t + o];
            red[64 + t] += red[64 + t + o];
            red[128 + t] += red[128 + t + o];
        }
        __syncthreads();
    }
    if (t == 0) {
        onull[tok] = red[0] + bnull[0];
        onum[tok]  = red[64] + bnum[0];
        obool[tok] = red[128] + bbool[0];
    }
    for (int o8 = 0; o8 < 8; o8++) {
        float a = 0.f;
        for (int d = t; d < 512; d += 64) a += row[d] * wts[d * 8 + o8];
        red[192 + t] = a;
        __syncthreads();
        for (int o = 32; o > 0; o >>= 1) {
            if (t < o) red[192 + t] += red[192 + t + o];
            __syncthreads();
        }
        if (t == 0) ots[tok * 8 + o8] = red[192] + bts[o8];
        __syncthreads();
    }
}

// ---------------------------------------------------------------------------
extern "C" void kernel_launch(void* const* d_in, const int* in_sizes, int n_in,
                              void* d_out, int out_size, void* d_ws, size_t ws_size,
                              hipStream_t stream) {
    (void)in_sizes; (void)out_size; (void)ws_size;

    const float* F[121];
    for (int i = 0; i < 121 && i < n_in; i++) F[i] = (const float*)d_in[i];
    const int* seq_row_ids = (const int*)d_in[106];
    const int* fk_adj      = (const int*)d_in[107];
    const int* is_padding  = (const int*)d_in[108];
    const int* column_ids  = (const int*)d_in[109];
    const int* sem_types   = (const int*)d_in[110];
    const int* is_null     = (const int*)d_in[111];
    const int* is_target   = (const int*)d_in[112];
    const float* num_vals  = (const float*)d_in[113];
    const float* ts_vals   = (const float*)d_in[114];
    const int* bool_vals   = (const int*)d_in[115];
    const int* cat_ids     = (const int*)d_in[116];
    const int* text_ids    = (const int*)d_in[117];
    const float* text_emb  = (const float*)d_in[118];
    const float* col_table = (const float*)d_in[119];
    const float* cat_table = (const float*)d_in[120];

    char* wsb = (char*)d_ws;
    float* h            = (float*)(wsb + (size_t)0);
    unsigned short* X   = (unsigned short*)(wsb + ((size_t)8  << 20));
    unsigned short* Qb  = (unsigned short*)(wsb + ((size_t)12 << 20));
    unsigned short* Kb  = (unsigned short*)(wsb + ((size_t)16 << 20));
    unsigned short* Vb  = (unsigned short*)(wsb + ((size_t)20 << 20));
    unsigned short* Vtb = (unsigned short*)(wsb + ((size_t)24 << 20));
    unsigned short* Ob  = (unsigned short*)(wsb + ((size_t)28 << 20));
    unsigned short* G   = (unsigned short*)(wsb + ((size_t)32 << 20));
    unsigned char* mo   = (unsigned char*)(wsb + ((size_t)48 << 20));
    unsigned char* mi   = (unsigned char*)(wsb + ((size_t)52 << 20));
    unsigned char* mc   = (unsigned char*)(wsb + ((size_t)56 << 20));
    unsigned short* Wbf = (unsigned short*)(wsb + ((size_t)60 << 20));

    float* out_h    = (float*)d_out;
    float* out_null = out_h + (size_t)MB * DD;
    float* out_num  = out_null + MB;
    float* out_bool = out_num + MB;
    float* out_ts   = out_bool + MB;
    float* out_cat  = out_ts + (size_t)MB * 8;

    // 1) value encode
    encode_kernel<<<MB, 128, 0, stream>>>(
        column_ids, sem_types, is_null, is_target, is_padding,
        num_vals, ts_vals, bool_vals, cat_ids, text_ids,
        text_emb, col_table, cat_table,
        F[4], F[3], F[17], F[101], F[100], F[105], F[104],
        F[0], F[2], F[1], F[103], F[102], F[99], F[98], F[16], h);

    // 2) masks
    build_masks_kernel<<<(BB * SS * SS) / 256, 256, 0, stream>>>(
        seq_row_ids, fk_adj, is_padding, column_ids, mo, mi, mc);

    const dim3 g512(32, 4);
    const dim3 g2048(32, 16);

    // 3) layers
    for (int l = 0; l < 4; l++) {
        const int base = 18 + 20 * l;
        // attention blocks in reference order: out, in, col
        const int normIdx[3] = {base + 13, base + 8, base + 2};
        const int wqIdx[3]   = {base + 16, base + 11, base + 5};
        const int wkIdx[3]   = {base + 14, base + 9,  base + 3};
        const int wvIdx[3]   = {base + 17, base + 12, base + 6};
        const int woIdx[3]   = {base + 15, base + 10, base + 4};
        const unsigned char* msk[3] = {mo, mi, mc};

        for (int a = 0; a < 3; a++) {
            rms_cast_kernel<<<MB, 128, 0, stream>>>(h, F[normIdx[a]], X, nullptr);

            cast_bf16_kernel<<<(DD * DD) / 256, 256, 0, stream>>>(F[wqIdx[a]], Wbf, DD * DD);
            gemm_bf16_t<FLAG_BF16><<<g512, 256, 0, stream>>>(
                X, Wbf, nullptr, nullptr, Qb, MB, DD, DD);
            cast_bf16_kernel<<<(DD * DD) / 256, 256, 0, stream>>>(F[wkIdx[a]], Wbf, DD * DD);
            gemm_bf16_t<FLAG_BF16><<<g512, 256, 0, stream>>>(
                X, Wbf, nullptr, nullptr, Kb, MB, DD, DD);
            cast_bf16_kernel<<<(DD * DD) / 256, 256, 0, stream>>>(F[wvIdx[a]], Wbf, DD * DD);
            gemm_bf16_t<FLAG_BF16><<<g512, 256, 0, stream>>>(
                X, Wbf, nullptr, nullptr, Vb, MB, DD, DD);

            transpose_v_kernel<<<(MB * DD) / 256, 256, 0, stream>>>(Vb, Vtb);

            attention_kernel<<<BB * HH * (SS / 16), 32, 0, stream>>>(Qb, Kb, Vtb, msk[a], Ob);

            cast_bf16_kernel<<<(DD * DD) / 256, 256, 0, stream>>>(F[woIdx[a]], Wbf, DD * DD);
            gemm_bf16_t<FLAG_F32 | FLAG_RES><<<g512, 256, 0, stream>>>(
                Ob, Wbf, nullptr, h, nullptr, MB, DD, DD);
        }

        // FFN
        rms_cast_kernel<<<MB, 128, 0, stream>>>(h, F[base + 7], X, nullptr);
        cast_bf16_kernel<<<(DD * DFF) / 256, 256, 0, stream>>>(F[base + 18], Wbf, DD * DFF);
        gemm_bf16_t<FLAG_GELU | FLAG_BF16 | FLAG_BIAS><<<g2048, 256, 0, stream>>>(
            X, Wbf, F[base + 0], nullptr, G, MB, DFF, DD);
        cast_bf16_kernel<<<(DFF * DD) / 256, 256, 0, stream>>>(F[base + 19], Wbf, DFF * DD);
        gemm_bf16_t<FLAG_F32 | FLAG_RES | FLAG_BIAS><<<g512, 256, 0, stream>>>(
            G, Wbf, F[base + 1], h, nullptr, MB, DD, DFF);
    }

    // 4) final norm: fp32 to output h, bf16 for decoder GEMM
    rms_cast_kernel<<<MB, 128, 0, stream>>>(h, F[15], X, out_h);

    // 5) small decoder heads
    decoders_kernel<<<MB, 64, 0, stream>>>(out_h,
        F[10], F[9], F[12], F[11], F[6], F[5], F[14], F[13],
        out_null, out_num, out_bool, out_ts);

    // 6) categorical decoder GEMM
    cast_bf16_kernel<<<(DD * DD) / 256, 256, 0, stream>>>(F[8], Wbf, DD * DD);
    gemm_bf16_t<FLAG_F32 | FLAG_BIAS><<<g512, 256, 0, stream>>>(
        X, Wbf, F[7], out_cat, nullptr, MB, DD, DD);
}
